// Mamba_CT_91070486544711
// MI455X (gfx1250) — compile-verified, matrix operand order checked by reading, not run on hardware
//
#include <hip/hip_runtime.h>
#include <hip/hip_bf16.h>

// ---------------------------------------------------------------------------
// CDNA5 / gfx1250: 3D-CNN stem (NDHWC) + 2x Mamba blocks + MLP head.
// GEMMs: v_wmma_f32_16x16x32_bf16. Big GEMMs stage 32xK weight tiles into LDS
// with the Tensor Data Mover (tensor_load_to_lds + s_wait_tensorcnt), shared
// by an 8-wave workgroup; each A fragment feeds 2 back-to-back WMMAs.
// ---------------------------------------------------------------------------

typedef __bf16 bf16_t;
typedef __attribute__((ext_vector_type(16))) __bf16 bf16x16;
typedef __attribute__((ext_vector_type(8)))  float  f32x8;
typedef __attribute__((ext_vector_type(4)))  unsigned int u32x4;
typedef __attribute__((ext_vector_type(8)))  unsigned int u32x8;

// ISA 7.12.2 fragment map: lane r (0-15) supplies row M (A) / col N (B); the
// 16 v16 elements cover K = {half*8+0..7, 16+half*8+0..7} -> two contiguous
// 8-float runs per lane, loaded as 4x b128.
__device__ __forceinline__ bf16x16 pack16(float4 p0, float4 p1, float4 p2,
                                          float4 p3, float msk) {
  bf16x16 v;
  v[0]  = (bf16_t)(p0.x * msk); v[1]  = (bf16_t)(p0.y * msk);
  v[2]  = (bf16_t)(p0.z * msk); v[3]  = (bf16_t)(p0.w * msk);
  v[4]  = (bf16_t)(p1.x * msk); v[5]  = (bf16_t)(p1.y * msk);
  v[6]  = (bf16_t)(p1.z * msk); v[7]  = (bf16_t)(p1.w * msk);
  v[8]  = (bf16_t)(p2.x * msk); v[9]  = (bf16_t)(p2.y * msk);
  v[10] = (bf16_t)(p2.z * msk); v[11] = (bf16_t)(p2.w * msk);
  v[12] = (bf16_t)(p3.x * msk); v[13] = (bf16_t)(p3.y * msk);
  v[14] = (bf16_t)(p3.z * msk); v[15] = (bf16_t)(p3.w * msk);
  return v;
}

__device__ __forceinline__ bf16x16 load_frag(const float* p, float msk) {
  float4 a0 = *(const float4*)(p);
  float4 a1 = *(const float4*)(p + 4);
  float4 a2 = *(const float4*)(p + 16);
  float4 a3 = *(const float4*)(p + 20);
  return pack16(a0, a1, a2, a3, msk);
}

// ---------------------------------------------------------------------------
// TDM: one contiguous nelem-float run (32 rows x K cols, ldw == K) -> LDS 0.
// D# per ISA ch.8: group0 {count=1 | lds_addr | global_addr | type=2},
// group1 {data_size=4B | tensor_dim0=nelem, dim1=1 | tile_dim0=nelem, dim1=1}.
// ---------------------------------------------------------------------------
__device__ __forceinline__ void tdm_load_tile(const float* gsrc, unsigned nelem) {
  unsigned long long ga = (unsigned long long)(const void*)gsrc;
  const unsigned galo = (unsigned)__builtin_amdgcn_readfirstlane((int)(unsigned)ga);
  const unsigned gahi = (unsigned)__builtin_amdgcn_readfirstlane((int)(unsigned)(ga >> 32));
  const unsigned ne   = (unsigned)__builtin_amdgcn_readfirstlane((int)nelem);
  u32x4 g0;
  g0[0] = 1u;                                   // count=1, user descriptor
  g0[1] = 0u;                                   // lds_addr = 0 (sW at offset 0)
  g0[2] = galo;                                 // global_addr[31:0]
  g0[3] = (gahi & 0x01FFFFFFu) | (2u << 30);    // global_addr[56:32] | type=2
  u32x8 g1;
  g1[0] = 0x20000u;                             // data_size = 2 -> 4 bytes
  g1[1] = (ne & 0xFFFFu) << 16;                 // tensor_dim0[15:0]
  g1[2] = ((ne >> 16) & 0xFFFFu) | (1u << 16);  // tensor_dim0[31:16] | dim1=1
  g1[3] = (ne & 0xFFFFu) << 16;                 // tile_dim0 = ne (<= 8192)
  g1[4] = 1u;                                   // tile_dim1 = 1, tile_dim2 = 0
  g1[5] = ne;                                   // tensor_dim0_stride
  g1[6] = 0u;
  g1[7] = 0u;
  asm volatile("tensor_load_to_lds %0, %1" :: "s"(g0), "s"(g1) : "memory");
}

// ---------------------------------------------------------------------------
// 8-wave WMMA GEMM: block computes 128 M-rows x 32 N-cols. Weight tile (32xK,
// contiguous, ldw==K) staged once via TDM into LDS; all 8 waves read B frags
// from LDS (ds_load_b128); each A fragment feeds 2 WMMAs. M%128==0, K%32==0,
// Npad%32==0 required. EP: 0 none, 1 relu(+bias), 2 softplus(+bias),
// 3 BN relu(x*scale+shift).
// ---------------------------------------------------------------------------
template <int K, int EP>
__global__ __launch_bounds__(256)
void gemm_tdm_kernel(const float* __restrict__ A, int lda,
                     const float* __restrict__ W,      // (Npad, K) contiguous
                     const float* __restrict__ bias,
                     const float* __restrict__ scale,
                     const float* __restrict__ shift,
                     float* __restrict__ out, int ldo,
                     int NT, int Nstore)
{
  __shared__ float sW[32 * K];

  const int bid  = blockIdx.x;
  const int mt8  = bid / NT;        // 128-row super tile
  const int nt   = bid % NT;        // 32-col super tile
  const int tid  = threadIdx.x;
  const int wave = tid >> 5;
  const int lane = tid & 31;
  const int half = lane >> 4;
  const int r    = lane & 15;

  if (wave == 0) {
    tdm_load_tile(W + (long)nt * 32 * K, 32u * K);
    __builtin_amdgcn_s_wait_tensorcnt(0);
  }
  __syncthreads();

  const float* arow  = A + (long)(mt8 * 128 + wave * 16 + r) * lda + half * 8;
  const float* brow0 = sW + (long)r * K + half * 8;          // N-tile 0
  const float* brow1 = sW + (long)(16 + r) * K + half * 8;   // N-tile 1

  f32x8 acc0 = {};
  f32x8 acc1 = {};
#pragma unroll
  for (int kb = 0; kb < K; kb += 32) {
    bf16x16 a  = load_frag(arow + kb, 1.f);
    bf16x16 b0 = load_frag(brow0 + kb, 1.f);
    bf16x16 b1 = load_frag(brow1 + kb, 1.f);
    acc0 = __builtin_amdgcn_wmma_f32_16x16x32_bf16(
        false, a, false, b0, (short)0, acc0, false, false);
    acc1 = __builtin_amdgcn_wmma_f32_16x16x32_bf16(
        false, a, false, b1, (short)0, acc1, false, false);
  }

#pragma unroll
  for (int j = 0; j < 2; ++j) {
    const f32x8& acc = (j == 0) ? acc0 : acc1;
    const int n = nt * 32 + j * 16 + r;
    float bval = 0.f, sc = 1.f, sh = 0.f;
    if (EP == 3) { sc = scale[n]; sh = shift[n]; }
    else if (bias != nullptr && n < Nstore) bval = bias[n];
#pragma unroll
    for (int v = 0; v < 8; ++v) {
      const int m = mt8 * 128 + wave * 16 + half * 8 + v;
      if (n < Nstore) {
        float x = acc[v];
        if (EP == 3)      x = fmaxf(x * sc + sh, 0.f);
        else {
          x += bval;
          if (EP == 1)      x = fmaxf(x, 0.f);
          else if (EP == 2) x = (x > 20.f) ? x : log1pf(__expf(x));
        }
        out[(long)m * ldo + n] = x;
      }
    }
  }
}

// ---------------------------------------------------------------------------
// Single-wave WMMA GEMM for tiny M (head). Same fragment scheme, no LDS.
// ---------------------------------------------------------------------------
template <int K, int EP>
__global__ __launch_bounds__(32)
void gemm_small_kernel(const float* __restrict__ A, int lda,
                       const float* __restrict__ W, int ldw,
                       const float* __restrict__ bias,
                       float* __restrict__ out, int ldo,
                       int NT, int Nstore)
{
  const int bid  = blockIdx.x;
  const int mt   = bid / NT;
  const int nt   = bid % NT;
  const int lane = threadIdx.x;
  const int half = lane >> 4;
  const int r    = lane & 15;

  const float* arow = A + (long)(mt * 16 + r) * lda + half * 8;
  const float* brow = W + (long)(nt * 16 + r) * ldw + half * 8;

  f32x8 acc = {};
#pragma unroll
  for (int kb = 0; kb < K; kb += 32) {
    bf16x16 a = load_frag(arow + kb, 1.f);
    bf16x16 b = load_frag(brow + kb, 1.f);
    acc = __builtin_amdgcn_wmma_f32_16x16x32_bf16(
        false, a, false, b, (short)0, acc, false, false);
  }

  const int n = nt * 16 + r;
  float bval = (bias != nullptr && n < Nstore) ? bias[n] : 0.f;
#pragma unroll
  for (int v = 0; v < 8; ++v) {
    const int m = mt * 16 + half * 8 + v;
    if (n < Nstore) {
      float x = acc[v] + bval;
      if (EP == 1) x = fmaxf(x, 0.f);
      out[(long)m * ldo + n] = x;
    }
  }
}

// ---------------------------------------------------------------------------
// WMMA implicit-GEMM conv, NDHWC, weights (tap,COUT,CIN). One wave computes
// 16 positions x NACC*16 couts: each A fragment feeds NACC back-to-back WMMAs.
// Spatial zero-pad = clamped address + x*msk (no divergence).
// ---------------------------------------------------------------------------
template <int CIN, int NACC>
__global__ __launch_bounds__(32)
void conv_wmma_kernel(const float* __restrict__ in,
                      const float* __restrict__ wp,   // (taps, COUT, CIN)
                      const float* __restrict__ bb,  const float* __restrict__ g,
                      const float* __restrict__ beta,const float* __restrict__ mean,
                      const float* __restrict__ var,
                      float* __restrict__ out,       // NDHWC
                      int COUT, int ID, int IH, int IW,
                      int OD, int OH, int OW,
                      int KS, int STR, int PAD, int NT)
{
  const int bid  = blockIdx.x;
  const int mt   = bid / NT;
  const int nt   = bid % NT;
  const int lane = threadIdx.x;
  const int half = lane >> 4;
  const int r    = lane & 15;

  const long m  = (long)mt * 16 + r;
  int  ow = (int)(m % OW); long t = m / OW;
  int  oh = (int)(t % OH); t /= OH;
  int  od = (int)(t % OD);
  int  nb = (int)(t / OD);

  const int ncout0 = nt * (NACC * 16) + r;
  const int taps   = KS * KS * KS;

  f32x8 acc[NACC] = {};
  for (int tap = 0; tap < taps; ++tap) {
    const int kd = tap / (KS * KS);
    const int kh = (tap / KS) % KS;
    const int kw = tap % KS;
    const int id = od * STR - PAD + kd;
    const int ih = oh * STR - PAD + kh;
    const int iw = ow * STR - PAD + kw;
    const bool ok = (unsigned)id < (unsigned)ID && (unsigned)ih < (unsigned)IH &&
                    (unsigned)iw < (unsigned)IW;
    const long sp  = ok ? ((((long)nb * ID + id) * IH + ih) * IW + iw) : 0;
    const float msk = ok ? 1.f : 0.f;

    const float* abase = in + sp * CIN + half * 8;
    const float* wtap  = wp + (long)tap * COUT * CIN + half * 8;
    __builtin_prefetch(abase + (long)IW * CIN, 0, 1);

#pragma unroll
    for (int cb = 0; cb < CIN; cb += 32) {
      bf16x16 a = load_frag(abase + cb, msk);
#pragma unroll
      for (int j = 0; j < NACC; ++j) {
        bf16x16 b = load_frag(wtap + (long)(ncout0 + j * 16) * CIN + cb, 1.f);
        acc[j] = __builtin_amdgcn_wmma_f32_16x16x32_bf16(
            false, a, false, b, (short)0, acc[j], false, false);
      }
    }
  }

#pragma unroll
  for (int j = 0; j < NACC; ++j) {
    const int co  = ncout0 + j * 16;
    const float inv = g[co] * rsqrtf(var[co] + 1e-5f);
    const float sh  = (bb[co] - mean[co]) * inv + beta[co];
#pragma unroll
    for (int v = 0; v < 8; ++v) {
      const long mo = (long)mt * 16 + half * 8 + v;
      out[mo * COUT + co] = fmaxf(acc[j][v] * inv + sh, 0.f);
    }
  }
}

// ---------------------------------------------------------------------------
// Direct conv for c1 (CIN=1, 3x3x3, s2 p1) + BN + ReLU, NDHWC output.
// ---------------------------------------------------------------------------
__global__ void conv1_direct_kernel(const float* __restrict__ in,
                                    const float* __restrict__ w,
                                    const float* __restrict__ bb,
                                    const float* __restrict__ g,
                                    const float* __restrict__ beta,
                                    const float* __restrict__ mean,
                                    const float* __restrict__ var,
                                    float* __restrict__ out,
                                    int Nb, int COUT,
                                    int ID, int IH, int IW,
                                    int OD, int OH, int OW)
{
  const long idx   = (long)blockIdx.x * blockDim.x + threadIdx.x;
  const long total = (long)Nb * OD * OH * OW * COUT;
  if (idx >= total) return;
  const int co = (int)(idx % COUT); long t = idx / COUT;
  const int ow = (int)(t % OW); t /= OW;
  const int oh = (int)(t % OH); t /= OH;
  const int od = (int)(t % OD);
  const int nb = (int)(t / OD);

  float s = 0.f;
#pragma unroll
  for (int kd = 0; kd < 3; ++kd)
#pragma unroll
    for (int kh = 0; kh < 3; ++kh)
#pragma unroll
      for (int kw = 0; kw < 3; ++kw) {
        const int id = od * 2 - 1 + kd;
        const int ih = oh * 2 - 1 + kh;
        const int iw = ow * 2 - 1 + kw;
        if (id < 0 || id >= ID || ih < 0 || ih >= IH || iw < 0 || iw >= IW)
          continue;
        s += w[co * 27 + kd * 9 + kh * 3 + kw] *
             in[((long)nb * ID + id) * IH * IW + (long)ih * IW + iw];
      }
  const float inv = g[co] * rsqrtf(var[co] + 1e-5f);
  out[idx] = fmaxf((s + bb[co] - mean[co]) * inv + beta[co], 0.f);
}

// Depthwise causal conv1d (k=4, left pad 3) + bias + SiLU.
__global__ void dwconv_silu_kernel(const float* __restrict__ xz,
                                   const float* __restrict__ cw,
                                   const float* __restrict__ cb,
                                   float* __restrict__ xt,
                                   int Bb, int L, int Dd)
{
  const long idx   = (long)blockIdx.x * blockDim.x + threadIdx.x;
  const long total = (long)Bb * L * Dd;
  if (idx >= total) return;
  const int d = (int)(idx % Dd);
  long t = idx / Dd;
  const int l = (int)(t % L);
  const int b = (int)(t / L);

  const float* xh = xz + (long)b * L * 512 + d;
  float s = cb[d];
#pragma unroll
  for (int j = 0; j < 4; ++j) {
    const int li = l - 3 + j;
    if (li >= 0) s += cw[d * 4 + j] * xh[(long)li * 512];
  }
  const float sig = 1.f / (1.f + __expf(-s));
  xt[idx] = s * sig;
}

// Selective scan: one block per batch; 256 threads = one channel each.
// dbc layout: (b,l,64): [0:8) dt_raw, [8:24) B, [24:40) C, rest zero.
__global__ __launch_bounds__(256)
void scan_kernel(const float* __restrict__ dt, const float* __restrict__ xt,
                 const float* __restrict__ dbc, const float* __restrict__ A_log,
                 float* __restrict__ ys, int L)
{
  __shared__ float sB[16];
  __shared__ float sC[16];
  const int b = blockIdx.x;
  const int d = threadIdx.x;

  float a[16], h[16];
#pragma unroll
  for (int s = 0; s < 16; ++s) {
    a[s] = -__expf(A_log[d * 16 + s]);
    h[s] = 0.f;
  }
  const long base = (long)b * L;
  for (int l = 0; l < L; ++l) {
    if (d < 16)      sB[d]      = dbc[(base + l) * 64 + 8 + d];
    else if (d < 32) sC[d - 16] = dbc[(base + l) * 64 + 24 + (d - 16)];
    __syncthreads();
    const float dtv = dt[(base + l) * 256 + d];
    const float xv  = xt[(base + l) * 256 + d];
    const float dx  = dtv * xv;
    float y = 0.f;
#pragma unroll
    for (int s = 0; s < 16; ++s) {
      h[s] = __expf(dtv * a[s]) * h[s] + dx * sB[s];
      y += h[s] * sC[s];
    }
    ys[(base + l) * 256 + d] = y;
    __syncthreads();
  }
}

// y = (ys + xt*D) * silu(z), z = xz cols 256..511. In-place into ys.
__global__ void ymix_kernel(float* __restrict__ ys, const float* __restrict__ xt,
                            const float* __restrict__ Dp,
                            const float* __restrict__ xz, long total)
{
  const long idx = (long)blockIdx.x * blockDim.x + threadIdx.x;
  if (idx >= total) return;
  const int d = (int)(idx % 256);
  const float zv  = xz[(idx / 256) * 512 + 256 + d];
  const float sig = 1.f / (1.f + __expf(-zv));
  const float y   = ys[idx] + xt[idx] * Dp[d];
  ys[idx] = y * zv * sig;
}

// Mean pool over L.
__global__ void pool_kernel(const float* __restrict__ seq,
                            float* __restrict__ pooled, int L)
{
  const int b = blockIdx.x;
  const int c = threadIdx.x;
  float s = 0.f;
  for (int l = 0; l < L; ++l) s += seq[((long)b * L + l) * 128 + c];
  pooled[b * 128 + c] = s * (1.f / (float)L);
}

// ---- small prep kernels -----------------------------------------------------
__global__ void repack_convw_kernel(const float* __restrict__ w,
                                    float* __restrict__ wp,
                                    int COUT, int CIN, int taps)
{
  const long idx   = (long)blockIdx.x * blockDim.x + threadIdx.x;
  const long total = (long)COUT * CIN * taps;
  if (idx >= total) return;
  const int tap = (int)(idx % taps); long t = idx / taps;
  const int c   = (int)(t % CIN);
  const int co  = (int)(t / CIN);
  wp[((long)tap * COUT + co) * CIN + c] = w[idx];
}

__global__ void pad_weight_kernel(const float* __restrict__ w,
                                  float* __restrict__ wp,
                                  int Nreal, int Npad, int Kreal, int Kpad)
{
  const long idx = (long)blockIdx.x * blockDim.x + threadIdx.x;
  if (idx >= (long)Npad * Kpad) return;
  const int n = (int)(idx / Kpad);
  const int k = (int)(idx % Kpad);
  wp[idx] = (n < Nreal && k < Kreal) ? w[(long)n * Kreal + k] : 0.f;
}

__global__ void bn_prep_kernel(const float* __restrict__ bb, const float* __restrict__ g,
                               const float* __restrict__ beta, const float* __restrict__ mean,
                               const float* __restrict__ var,
                               float* __restrict__ scale, float* __restrict__ shift, int C)
{
  const int c = blockIdx.x * blockDim.x + threadIdx.x;
  if (c >= C) return;
  const float inv = g[c] * rsqrtf(var[c] + 1e-5f);
  scale[c] = inv;
  shift[c] = (bb[c] - mean[c]) * inv + beta[c];
}

// ---------------------------------------------------------------------------
// Host orchestration
// ---------------------------------------------------------------------------
namespace {

struct MambaPtrs {
  const float *in_proj, *conv_w, *conv_b, *xproj, *dtw, *dtb, *A_log, *D, *outw;
};

constexpr int  kB    = 16;
constexpr int  kL    = 2048;
constexpr int  kDI   = 256;
constexpr long kMtok = (long)kB * kL;   // 32768

inline int cdiv(long a, long b) { return (int)((a + b - 1) / b); }

template <int K, int EP>
void launch_gemm(const float* A, int lda, const float* W,
                 const float* bias, const float* scale, const float* shift,
                 float* out, int ldo, int M, int Npad, int Nstore, hipStream_t s)
{
  const int NT = Npad / 32;
  gemm_tdm_kernel<K, EP><<<(M / 128) * NT, 256, 0, s>>>(A, lda, W, bias, scale,
                                                        shift, out, ldo, NT, Nstore);
}

void run_mamba(const float* seq_in, float* seq_out, const MambaPtrs& p,
               float* xz, float* xt, float* dt, float* dbc, float* ys,
               float* xprojp, float* dtwp, hipStream_t s)
{
  const int M = (int)kMtok;
  // pad xproj (40,256)->(64,256) and dtw (256,8)->(256,32) [zero-filled]
  pad_weight_kernel<<<cdiv(64 * 256, 256), 256, 0, s>>>(p.xproj, xprojp, 40, 64, 256, 256);
  pad_weight_kernel<<<cdiv(256 * 32, 256), 256, 0, s>>>(p.dtw, dtwp, 256, 256, 8, 32);

  launch_gemm<128, 0>(seq_in, 128, p.in_proj, nullptr, nullptr, nullptr,
                      xz, 512, M, 512, 512, s);
  dwconv_silu_kernel<<<cdiv(kMtok * kDI, 256), 256, 0, s>>>(xz, p.conv_w,
                                                            p.conv_b, xt, kB, kL, kDI);
  // xproj -> dbc (M,64); cols 40..63 are exact zeros (zero weight rows)
  launch_gemm<256, 0>(xt, 256, xprojp, nullptr, nullptr, nullptr,
                      dbc, 64, M, 64, 64, s);
  // dt = softplus(dbc[:,0:32] x (256,32)^T + dtb); padded dtw zeros cols 8..31
  launch_gemm<32, 2>(dbc, 64, dtwp, p.dtb, nullptr, nullptr,
                     dt, 256, M, 256, 256, s);
  scan_kernel<<<kB, 256, 0, s>>>(dt, xt, dbc, p.A_log, ys, kL);
  ymix_kernel<<<cdiv(kMtok * kDI, 256), 256, 0, s>>>(ys, xt, p.D, xz, kMtok * kDI);
  launch_gemm<256, 0>(ys, 256, p.outw, nullptr, nullptr, nullptr,
                      seq_out, 128, M, 128, 128, s);
}

} // namespace

extern "C" void kernel_launch(void* const* d_in, const int* in_sizes, int n_in,
                              void* d_out, int out_size, void* d_ws, size_t ws_size,
                              hipStream_t stream)
{
  (void)in_sizes; (void)n_in; (void)out_size; (void)ws_size;

  const float* x = (const float*)d_in[0];
  const float** c1 = (const float**)&d_in[1];
  const float** c2 = (const float**)&d_in[7];
  const float** c3 = (const float**)&d_in[13];
  const float** cp = (const float**)&d_in[19];
  auto mk = [&](int base) {
    MambaPtrs m;
    m.in_proj = (const float*)d_in[base + 0];
    m.conv_w  = (const float*)d_in[base + 1];
    m.conv_b  = (const float*)d_in[base + 2];
    m.xproj   = (const float*)d_in[base + 3];
    m.dtw     = (const float*)d_in[base + 4];
    m.dtb     = (const float*)d_in[base + 5];
    m.A_log   = (const float*)d_in[base + 6];
    m.D       = (const float*)d_in[base + 7];
    m.outw    = (const float*)d_in[base + 8];
    return m;
  };
  MambaPtrs m1 = mk(25), m2 = mk(34);
  const float* hw1 = (const float*)d_in[43];
  const float* hb1 = (const float*)d_in[44];
  const float* hw2 = (const float*)d_in[45];
  const float* hb2 = (const float*)d_in[46];
  float* out = (float*)d_out;

  // ---- workspace (floats) ----
  float* ws = (float*)d_ws;
  const size_t off_h1   = 0;                    // 16,777,216 NDHWC (alias xz)
  const size_t off_h2   = off_h1 + 16777216;    // 16,777,216 NDHWC (alias xt+dt)
  const size_t off_h3   = off_h2 + 16777216;    //  4,194,304 NDHWC == (B,L,128)
  const size_t off_seqA = off_h3 + 4194304;
  const size_t off_seqB = off_seqA + 4194304;
  const size_t off_dbc  = off_seqB + 4194304;   //  2,097,152 (M x 64)
  const size_t off_ys   = off_dbc + 2097152;    //  8,388,608
  const size_t off_pool = off_ys + 8388608;     //  2,048
  const size_t off_hid  = off_pool + 2048;      //  1,024
  const size_t off_wp2  = off_hid + 1024;       //  110,592  (27,64,64)
  const size_t off_wp3  = off_wp2 + 110592;     //  221,184  (27,128,64)
  const size_t off_xpp  = off_wp3 + 221184;     //  16,384   (64,256)
  const size_t off_dtp  = off_xpp + 16384;      //  8,192    (256,32)
  const size_t off_w2p  = off_dtp + 8192;       //  1,024    (16,64)
  const size_t off_bns  = off_w2p + 1024;       //  128
  const size_t off_bnh  = off_bns + 128;        //  128

  float* h1   = ws + off_h1;
  float* h2   = ws + off_h2;
  float* h3   = ws + off_h3;
  float* seqA = ws + off_seqA;
  float* seqB = ws + off_seqB;
  float* xz   = ws + off_h1;
  float* xt   = ws + off_h2;
  float* dt   = ws + off_h2 + 8388608;
  float* dbc  = ws + off_dbc;
  float* ys   = ws + off_ys;
  float* pool = ws + off_pool;
  float* hid  = ws + off_hid;
  float* wp2  = ws + off_wp2;
  float* wp3  = ws + off_wp3;
  float* xpp  = ws + off_xpp;
  float* dtp  = ws + off_dtp;
  float* w2p  = ws + off_w2p;
  float* bns  = ws + off_bns;
  float* bnh  = ws + off_bnh;

  // ---- weight prep ----
  repack_convw_kernel<<<cdiv(110592, 256), 256, 0, stream>>>(c2[0], wp2, 64, 64, 27);
  repack_convw_kernel<<<cdiv(221184, 256), 256, 0, stream>>>(c3[0], wp3, 128, 64, 27);
  pad_weight_kernel<<<cdiv(16 * 64, 256), 256, 0, stream>>>(hw2, w2p, 2, 16, 64, 64);
  bn_prep_kernel<<<2, 64, 0, stream>>>(cp[1], cp[2], cp[3], cp[4], cp[5], bns, bnh, 128);

  // ---- stem (all NDHWC) ----
  conv1_direct_kernel<<<cdiv((long)kB * 64 * 16 * 32 * 32, 256), 256, 0, stream>>>(
      x, c1[0], c1[1], c1[2], c1[3], c1[4], c1[5],
      h1, kB, 64, 32, 64, 64, 16, 32, 32);
  // c2: 64->64, k3 s1 p1; one wave = 16 pos x 64 couts
  {
    const long M = (long)kB * 16 * 32 * 32;
    conv_wmma_kernel<64, 4><<<(int)(M / 16), 32, 0, stream>>>(
        h1, wp2, c2[1], c2[2], c2[3], c2[4], c2[5], h2,
        64, 16, 32, 32, 16, 32, 32, 3, 1, 1, 1);
  }
  // c3: 64->128, k3 s2 p1 -> h3 == (B,L,128); one wave = 16 pos x 64 couts
  {
    const long M = (long)kB * 8 * 16 * 16;
    conv_wmma_kernel<64, 4><<<(int)(M / 16) * 2, 32, 0, stream>>>(
        h2, wp3, c3[1], c3[2], c3[3], c3[4], c3[5], h3,
        128, 16, 32, 32, 8, 16, 16, 3, 2, 1, 2);
  }
  // p: 1x1x1 conv == GEMM with BN+ReLU epilogue -> seqA
  launch_gemm<128, 3>(h3, 128, cp[0], nullptr, bns, bnh,
                      seqA, 128, (int)kMtok, 128, 128, stream);

  // ---- Mamba blocks ----
  run_mamba(seqA, seqB, m1, xz, xt, dt, dbc, ys, xpp, dtp, stream);
  run_mamba(seqB, seqA, m2, xz, xt, dt, dbc, ys, xpp, dtp, stream);

  // ---- head ----
  pool_kernel<<<kB, 128, 0, stream>>>(seqA, pool, kL);
  gemm_small_kernel<128, 1><<<4, 32, 0, stream>>>(pool, 128, hw1, 128, hb1,
                                                  hid, 64, 4, 64);
  gemm_small_kernel<64, 0><<<1, 32, 0, stream>>>(hid, 64, w2p, 64, hb2,
                                                 out, 2, 1, 2);
}
